// LSTM_75548474737118
// MI455X (gfx1250) — matrix operand order repaired and checked by reading, not compiled, not run on hardware
//
#include <hip/hip_runtime.h>
#include <math.h>

// ---------------- types ----------------
typedef __attribute__((ext_vector_type(16))) __bf16 v16bf;
typedef __attribute__((ext_vector_type(8)))  float  v8f;
typedef __attribute__((ext_vector_type(4)))  unsigned int u32x4;
typedef __attribute__((ext_vector_type(8)))  int i32x8;
typedef __attribute__((ext_vector_type(4)))  int i32x4;

union FragAB { uint4 u[2]; v16bf v; };

__device__ __forceinline__ unsigned short f2bf(float f) {
  unsigned u = __builtin_bit_cast(unsigned, f);
  unsigned r = ((u >> 16) & 1u) + 0x7fffu;   // round-to-nearest-even
  return (unsigned short)((u + r) >> 16);
}

__device__ __forceinline__ float sigmoidf_(float x) {
  return 1.0f / (1.0f + __expf(-x));
}

// CDNA5 async global->LDS copy (ASYNCcnt-tracked), 16B per lane.
__device__ __forceinline__ void async_copy16(unsigned lds_off, const void* g) {
  asm volatile("global_load_async_to_lds_b128 %0, %1, off"
               :: "v"(lds_off), "v"((unsigned long long)g)
               : "memory");
}
__device__ __forceinline__ void wait_async0() {
  asm volatile("s_wait_asynccnt 0x0" ::: "memory");
}

// Problem constants
#define BATCH 64
#define SEQ   1024
#define IDIM  512
#define HDIM  512
#define G4    2048          // 4*H
#define CHUNK 32            // hidden units per persistent WG
#define NWG   16            // 512 / CHUNK
#define NSL   128           // gate rows per WG = 4*CHUNK
#define OUTH  33554432L     // B*S*H

// ---------------------------------------------------------------------------
// Kernel 1: permute+convert weights to bf16, build combined bias.
//   r' = (hid/32)*128 + g*32 + (hid%32)
// ---------------------------------------------------------------------------
__global__ __launch_bounds__(256) void convert_weights(
    const float* __restrict__ wii, const float* __restrict__ wif,
    const float* __restrict__ wig, const float* __restrict__ wio,
    const float* __restrict__ whi, const float* __restrict__ whf,
    const float* __restrict__ whg, const float* __restrict__ who,
    const float* __restrict__ bii, const float* __restrict__ bif,
    const float* __restrict__ big, const float* __restrict__ bio,
    const float* __restrict__ bhi, const float* __restrict__ bhf,
    const float* __restrict__ bhg, const float* __restrict__ bho,
    unsigned short* __restrict__ Wi, unsigned short* __restrict__ Wh,
    float* __restrict__ bias)
{
  int t = blockIdx.x * 256 + threadIdx.x;       // over 2048*512
  int r   = t >> 9;
  int col = t & 511;
  int g   = r >> 9;
  int hid = r & 511;
  int w   = hid >> 5;
  int j   = hid & 31;
  int rp  = w * NSL + g * CHUNK + j;

  const float* wi = (g == 0) ? wii : (g == 1) ? wif : (g == 2) ? wig : wio;
  const float* wh = (g == 0) ? whi : (g == 1) ? whf : (g == 2) ? whg : who;
  Wi[(long)rp * 512 + col] = f2bf(wi[hid * 512 + col]);
  Wh[(long)rp * 512 + col] = f2bf(wh[hid * 512 + col]);

  if (col == 0) {
    const float* bi = (g == 0) ? bii : (g == 1) ? bif : (g == 2) ? big : bio;
    const float* bh = (g == 0) ? bhi : (g == 1) ? bhf : (g == 2) ? bhg : bho;
    bias[rp] = bi[hid] + bh[hid];
  }
}

// ---------------------------------------------------------------------------
// Kernel 2: init bf16 h buffer (= h0) and zero the grid-barrier words.
// ---------------------------------------------------------------------------
__global__ __launch_bounds__(256) void init_state(
    const float* __restrict__ h0, unsigned short* __restrict__ hbufb,
    unsigned* __restrict__ bar)
{
  int gid = blockIdx.x * 256 + threadIdx.x;     // 32768 threads
  hbufb[gid] = f2bf(h0[gid]);
  if (gid < 8) bar[gid] = 0u;
}

// ---------------------------------------------------------------------------
// Kernel 3: x_proj GEMM.  C(65536 x 2048) = Xf32(65536 x 512) * Wi_bf16^T
// B tile staged with CDNA5 async global->LDS copies.
// ---------------------------------------------------------------------------
__global__ __launch_bounds__(256) void xproj_gemm(
    const float* __restrict__ inp,              // (B,S,I) row = b*S+s
    const unsigned short* __restrict__ Wi,      // (2048 x 512) bf16 permuted
    float* __restrict__ xproj)                  // (S, B, 2048) f32
{
  __shared__ unsigned short Al[128][48];
  __shared__ unsigned short Bl[128][48];

  const int tid   = threadIdx.x;
  const int lane  = tid & 31;
  const int wave  = tid >> 5;
  const int waveM = wave & 1;
  const int waveN = wave >> 1;
  const int half  = lane >> 4;
  const int l16   = lane & 15;
  const long rowBase = (long)blockIdx.x * 128;
  const int  colBase = blockIdx.y * 128;

  v8f acc[4][2];
#pragma unroll
  for (int i = 0; i < 4; ++i)
#pragma unroll
    for (int j = 0; j < 2; ++j) acc[i][j] = (v8f)0.0f;

  for (int kc = 0; kc < 512; kc += 32) {
    __syncthreads();
    // stage B tile via async copies (bf16 weights, raw bytes)
#pragma unroll
    for (int p = 0; p < 2; ++p) {
      int row = p * 64 + (tid >> 2);
      int c8  = (tid & 3) * 8;
      async_copy16((unsigned)(unsigned long long)&Bl[row][c8],
                   Wi + (long)(colBase + row) * 512 + kc + c8);
    }
    // stage A tile (f32 -> bf16 conversion, manual)
#pragma unroll
    for (int p = 0; p < 4; ++p) {
      int row = p * 32 + (tid >> 3);
      int c4  = (tid & 7) * 4;
      float4 f = *(const float4*)(inp + (rowBase + row) * 512L + kc + c4);
      unsigned short* d = &Al[row][c4];
      d[0] = f2bf(f.x); d[1] = f2bf(f.y); d[2] = f2bf(f.z); d[3] = f2bf(f.w);
    }
    wait_async0();
    __syncthreads();

#pragma unroll
    for (int nt = 0; nt < 2; ++nt) {
      FragAB bf_;
      const unsigned short* bb = &Bl[waveN * 32 + nt * 16 + l16][half * 8];
      bf_.u[0] = *(const uint4*)(bb);
      bf_.u[1] = *(const uint4*)(bb + 16);
#pragma unroll
      for (int mt = 0; mt < 4; ++mt) {
        FragAB af;
        const unsigned short* ab = &Al[waveM * 64 + mt * 16 + l16][half * 8];
        af.u[0] = *(const uint4*)(ab);
        af.u[1] = *(const uint4*)(ab + 16);
        acc[mt][nt] = __builtin_amdgcn_wmma_f32_16x16x32_bf16(
            false, af.v, false, bf_.v, (short)0, acc[mt][nt], false, false);
      }
    }
  }

  // store: row r = b*1024+s  ->  xproj[(s*64+b)*2048 + col]
#pragma unroll
  for (int mt = 0; mt < 4; ++mt)
#pragma unroll
    for (int nt = 0; nt < 2; ++nt)
#pragma unroll
      for (int v = 0; v < 8; ++v) {
        long r = rowBase + waveM * 64 + mt * 16 + v + 8 * half;
        int  b = (int)(r >> 10);
        int  s = (int)(r & 1023);
        int  col = colBase + waveN * 32 + nt * 16 + l16;
        xproj[((long)s * 64 + b) * 2048 + col] = acc[mt][nt][v];
      }
}

// ---------------------------------------------------------------------------
// Kernel 4: persistent recurrent scan. 16 WGs x 256 threads (8 waves).
// Wh slice pinned in LDS via the Tensor Data Mover (with row padding);
// per-step h exchange in bf16 via async global->LDS copies.
// ---------------------------------------------------------------------------
__global__ __launch_bounds__(256) void lstm_scan(
    const unsigned short* __restrict__ Wh,   // (2048 x 512) bf16 permuted
    const float* __restrict__ bias,          // (2048) permuted
    const float* __restrict__ xproj,         // (S, B, 2048)
    const float* __restrict__ c0,            // (B, H)
    unsigned short* __restrict__ hbufb,      // 2 x (B, H) bf16
    unsigned* __restrict__ bar,
    float* __restrict__ out)                 // B*S*H + h_f + c_f
{
  __shared__ unsigned short WhL[128][528];
  __shared__ unsigned short hL[64][528];
  __shared__ float gL[64][132];
  __shared__ float biasL[128];

  const int tid   = threadIdx.x;
  const int w     = blockIdx.x;              // 0..15
  const int lane  = tid & 31;
  const int wave  = tid >> 5;
  const int waveM = wave & 1;                // 2 waves in M (32 rows each)
  const int waveN = wave >> 1;               // 4 waves in N (32 cols each)
  const int half  = lane >> 4;
  const int l16   = lane & 15;

  // ---- pin Wh slice in LDS via TDM ----
  // tile: 512 x 128 elements of 2B, row stride 512; LDS pad: after each
  // 1024B row (256 DWORDs, code 7) insert 8 DWORDs (code 7) -> stride 528.
  if (wave == 0) {
    unsigned ldsbase = (unsigned)(unsigned long long)(&WhL[0][0]);
    unsigned long long ga =
        (unsigned long long)(Wh + (long)w * NSL * 512);
    u32x4 g0;
    g0.x = 1u;                                            // count=1
    g0.y = ldsbase;                                       // lds_addr
    g0.z = (unsigned)(ga & 0xFFFFFFFFull);                // global_addr lo
    g0.w = (unsigned)((ga >> 32) & 0x1FFFFFFull) | (2u << 30);  // type=2
    i32x8 g1;
    g1[0] = (1 << 16) | (1 << 20) | (7 << 22) | (7 << 25);  // 2B, pad 256DW/8DW
    g1[1] = (int)(512u << 16);        // tensor_dim0[15:0] << 16
    g1[2] = (int)(2048u << 16);       // tensor_dim1[15:0] << 16
    g1[3] = (int)(512u << 16);        // tile_dim0 << 16
    g1[4] = 128;                      // tile_dim1
    g1[5] = 512;                      // tensor_dim0_stride lo
    g1[6] = 0;
    g1[7] = 0;
    i32x4 z4 = {0, 0, 0, 0};
    i32x8 z8 = {0, 0, 0, 0, 0, 0, 0, 0};
    __builtin_amdgcn_tensor_load_to_lds(g0, g1, z4, z4, z8, 0);
    __builtin_amdgcn_s_wait_tensorcnt(0);
  }
  if (tid < 128) biasL[tid] = bias[w * NSL + tid];

  // cell state in registers: element e = tid + k*256 -> (b=e/32, j=e%32)
  float c_reg[8];
#pragma unroll
  for (int k = 0; k < 8; ++k) {
    int e = tid + k * 256;
    int b = e >> 5, j = e & 31;
    c_reg[k] = c0[b * 512 + w * CHUNK + j];
  }
  __syncthreads();

  int cur = 0;
  for (int t = 0; t < SEQ; ++t) {
    // stage h (bf16 global -> LDS) via async copies; 16 x 16B per thread
    const unsigned short* hsrc = hbufb + cur * 32768;
    for (int i = tid; i < 4096; i += 256) {
      int idx = i * 8;               // element index in 64x512
      int row = idx >> 9;
      int col = idx & 511;
      async_copy16((unsigned)(unsigned long long)&hL[row][col], hsrc + idx);
    }
    // prefetch next step's x_proj slice (64 x 128 f32, 128B per thread)
    if (t + 1 < SEQ) {
      const float* xnext = xproj + ((long)(t + 1) * 64) * 2048 + w * NSL;
      __builtin_prefetch(xnext + (tid >> 2) * 2048 + (tid & 3) * 32, 0, 3);
    }
    wait_async0();
    __syncthreads();

    // gates(64x128) = hL(64x512) x WhL^T
    v8f acc[2][2];
#pragma unroll
    for (int i = 0; i < 2; ++i)
#pragma unroll
      for (int j = 0; j < 2; ++j) acc[i][j] = (v8f)0.0f;

    for (int kc = 0; kc < 512; kc += 32) {
#pragma unroll
      for (int nt = 0; nt < 2; ++nt) {
        FragAB bf_;
        const unsigned short* bb = &WhL[waveN * 32 + nt * 16 + l16][kc + half * 8];
        bf_.u[0] = *(const uint4*)(bb);
        bf_.u[1] = *(const uint4*)(bb + 16);
#pragma unroll
        for (int mt = 0; mt < 2; ++mt) {
          FragAB af;
          const unsigned short* ab = &hL[waveM * 32 + mt * 16 + l16][kc + half * 8];
          af.u[0] = *(const uint4*)(ab);
          af.u[1] = *(const uint4*)(ab + 16);
          acc[mt][nt] = __builtin_amdgcn_wmma_f32_16x16x32_bf16(
              false, af.v, false, bf_.v, (short)0, acc[mt][nt], false, false);
        }
      }
    }

    // spill gates to LDS for cross-wave gate gather
#pragma unroll
    for (int mt = 0; mt < 2; ++mt)
#pragma unroll
      for (int nt = 0; nt < 2; ++nt)
#pragma unroll
        for (int v = 0; v < 8; ++v)
          gL[waveM * 32 + mt * 16 + v + 8 * half][waveN * 32 + nt * 16 + l16] =
              acc[mt][nt][v];
    __syncthreads();

    // elementwise cell update
    const float* xp = xproj + ((long)t * 64) * 2048 + w * NSL;
    unsigned short* hdst = hbufb + (cur ^ 1) * 32768;
#pragma unroll
    for (int k = 0; k < 8; ++k) {
      int e = tid + k * 256;
      int b = e >> 5, j = e & 31;
      const float* xr = xp + (long)b * 2048;
      float gi = gL[b][j]      + biasL[j]      + xr[j];
      float gf = gL[b][32 + j] + biasL[32 + j] + xr[32 + j];
      float gg = gL[b][64 + j] + biasL[64 + j] + xr[64 + j];
      float go = gL[b][96 + j] + biasL[96 + j] + xr[96 + j];
      float it = sigmoidf_(gi);
      float ft = sigmoidf_(gf);
      float gt = tanhf(gg);
      float ot = sigmoidf_(go);
      float c  = ft * c_reg[k] + it * gt;
      c_reg[k] = c;
      float h  = ot * tanhf(c);
      hdst[b * 512 + w * CHUNK + j] = f2bf(h);
      out[((long)b * SEQ + t) * 512 + w * CHUNK + j] = h;
      if (t == SEQ - 1) {
        out[OUTH + b * 512 + w * CHUNK + j]         = h;
        out[OUTH + 32768 + b * 512 + w * CHUNK + j] = c;
      }
    }
    __syncthreads();

    // device-scope barrier (ticket + generation)
    __threadfence();
    if (tid == 0) {
      unsigned g0 = __hip_atomic_load(bar + 1, __ATOMIC_ACQUIRE,
                                      __HIP_MEMORY_SCOPE_AGENT);
      unsigned tk = __hip_atomic_fetch_add(bar, 1u, __ATOMIC_ACQ_REL,
                                           __HIP_MEMORY_SCOPE_AGENT);
      if (tk == NWG - 1) {
        __hip_atomic_store(bar, 0u, __ATOMIC_RELAXED, __HIP_MEMORY_SCOPE_AGENT);
        __hip_atomic_fetch_add(bar + 1, 1u, __ATOMIC_ACQ_REL,
                               __HIP_MEMORY_SCOPE_AGENT);
      } else {
        while (__hip_atomic_load(bar + 1, __ATOMIC_ACQUIRE,
                                 __HIP_MEMORY_SCOPE_AGENT) == g0) {
          __builtin_amdgcn_s_sleep(8);
        }
      }
    }
    __syncthreads();
    __threadfence();
    cur ^= 1;
  }
}

// ---------------------------------------------------------------------------
extern "C" void kernel_launch(void* const* d_in, const int* in_sizes, int n_in,
                              void* d_out, int out_size, void* d_ws,
                              size_t ws_size, hipStream_t stream) {
  const float* inputs = (const float*)d_in[0];
  const float* h0     = (const float*)d_in[1];
  const float* c0     = (const float*)d_in[2];
  const float* w_ii = (const float*)d_in[3];
  const float* w_if = (const float*)d_in[4];
  const float* w_ig = (const float*)d_in[5];
  const float* w_io = (const float*)d_in[6];
  const float* w_hi = (const float*)d_in[7];
  const float* w_hf = (const float*)d_in[8];
  const float* w_hg = (const float*)d_in[9];
  const float* w_ho = (const float*)d_in[10];
  const float* b_ii = (const float*)d_in[11];
  const float* b_if = (const float*)d_in[12];
  const float* b_ig = (const float*)d_in[13];
  const float* b_io = (const float*)d_in[14];
  const float* b_hi = (const float*)d_in[15];
  const float* b_hf = (const float*)d_in[16];
  const float* b_ho = (const float*)d_in[17];   // dict order: b_ho before b_hg
  const float* b_hg = (const float*)d_in[18];

  char* ws = (char*)d_ws;
  unsigned short* Wi   = (unsigned short*)(ws + 0);          // 2 MB
  unsigned short* Wh   = (unsigned short*)(ws + 2097152);    // 2 MB
  float*          bias = (float*)(ws + 4194304);             // 8 KB
  unsigned short* hbufb= (unsigned short*)(ws + 4202496);    // 128 KB (x2 buf)
  unsigned*       bar  = (unsigned*)(ws + 4464640);          // 256 B
  float*          xprj = (float*)(ws + 4464896);             // 512 MB

  convert_weights<<<4096, 256, 0, stream>>>(
      w_ii, w_if, w_ig, w_io, w_hi, w_hf, w_hg, w_ho,
      b_ii, b_if, b_ig, b_io, b_hi, b_hf, b_hg, b_ho, Wi, Wh, bias);

  init_state<<<128, 256, 0, stream>>>(h0, hbufb, bar);

  xproj_gemm<<<dim3(512, 16), 256, 0, stream>>>(inputs, Wi, xprj);

  lstm_scan<<<NWG, 256, 0, stream>>>(Wh, bias, xprj, c0, hbufb, bar,
                                     (float*)d_out);
}
